// AttentionBlock_31885837205534
// MI455X (gfx1250) — matrix operand order repaired and checked by reading, not compiled, not run on hardware
//
#include <hip/hip_runtime.h>

typedef __bf16 v16bf __attribute__((ext_vector_type(16)));
typedef float  v8f   __attribute__((ext_vector_type(8)));
typedef int    v4i   __attribute__((ext_vector_type(4)));

#define B_    8
#define C_    1024
#define L_    1024
#define H_    16
#define CH_   64
#define GROUPS_ 32

typedef __attribute__((address_space(1))) v4i* gv4i_p;  // global int4*
typedef __attribute__((address_space(3))) v4i* lv4i_p;  // LDS int4*

// Async global->LDS 16-byte copy (CDNA5 GLOBAL_LOAD_ASYNC_TO_LDS_B128, ASYNCcnt).
static __device__ inline void async_copy_b128(const void* g, void* l) {
#if __has_builtin(__builtin_amdgcn_global_load_async_to_lds_b128)
  __builtin_amdgcn_global_load_async_to_lds_b128((gv4i_p)g, (lv4i_p)l, 0, 0);
#else
  unsigned loff = (unsigned)(unsigned long long)l;
  asm volatile("global_load_async_to_lds_b128 %0, %1, off" ::"v"(loff), "v"(g)
               : "memory");
#endif
}

static __device__ inline void wait_async0() {
#if __has_builtin(__builtin_amdgcn_s_wait_asynccnt)
  __builtin_amdgcn_s_wait_asynccnt(0);
#else
  asm volatile("s_wait_asynccnt 0" ::: "memory");
#endif
}

static __device__ inline v8f wmma_bf16(v16bf a, v16bf b, v8f c) {
  return __builtin_amdgcn_wmma_f32_16x16x32_bf16(false, a, false, b, (short)0, c,
                                                 false, false);
}

// A operand (16x32, K-contraction): lane l -> M = l%16, K groups {8h..8h+7, 16+8h..+7}
// Storage assumed row-major [M][K] with K contiguous.
static __device__ inline v16bf load_fragA(const __bf16* base, int pitch) {
  int l = threadIdx.x & 31;
  const __bf16* p = base + (l & 15) * pitch + 8 * (l >> 4);
  union { uint4 u[2]; v16bf v; } U;
  U.u[0] = *(const uint4*)(p);
  U.u[1] = *(const uint4*)(p + 16);
  return U.v;
}

// B operand (32x16): lane l -> N = l%16, K groups {8h..8h+7, 16+8h..+7}
// Storage assumed [N][K] with K contiguous.
static __device__ inline v16bf load_fragB(const __bf16* base, int pitch) {
  int l = threadIdx.x & 31;
  const __bf16* p = base + (l & 15) * pitch + 8 * (l >> 4);
  union { uint4 u[2]; v16bf v; } U;
  U.u[0] = *(const uint4*)(p);
  U.u[1] = *(const uint4*)(p + 16);
  return U.v;
}

static __device__ inline unsigned pack_bf16(float a, float b) {
  union { __bf16 h[2]; unsigned u; } U;
  U.h[0] = (__bf16)a; U.h[1] = (__bf16)b;
  return U.u;
}

// ---------------- fp32 -> bf16 weight conversion (one-shot, tiny) ----------------
__global__ __launch_bounds__(256) void cvt_bf16_kernel(const float* __restrict__ src,
                                                       __bf16* __restrict__ dst) {
  int i = blockIdx.x * 256 + threadIdx.x;
  float4 v = ((const float4*)src)[i];
  union { __bf16 h[4]; uint2 u; } U;
  U.h[0] = (__bf16)v.x; U.h[1] = (__bf16)v.y;
  U.h[2] = (__bf16)v.z; U.h[3] = (__bf16)v.w;
  ((uint2*)dst)[i] = U.u;
}

// ---------------- GroupNorm (memory bound, float4 path) ----------------
__global__ __launch_bounds__(256) void groupnorm_kernel(
    const float* __restrict__ x, const float* __restrict__ w,
    const float* __restrict__ bb, float* __restrict__ xn) {
  __shared__ float r1[256], r2[256];
  int tid = threadIdx.x;
  int b = blockIdx.x >> 5, g = blockIdx.x & 31;
  const float4* xb4 = (const float4*)(x + ((size_t)b * C_ + g * 32) * L_);
  float4* xo4 = (float4*)(xn + ((size_t)b * C_ + g * 32) * L_);
  float s = 0.f, ss = 0.f;
#pragma unroll 4
  for (int i = 0; i < 32; ++i) {
    float4 v = xb4[tid + 256 * i];
    s += v.x + v.y + v.z + v.w;
    ss += v.x * v.x + v.y * v.y + v.z * v.z + v.w * v.w;
  }
  r1[tid] = s; r2[tid] = ss;
  __syncthreads();
  for (int st = 128; st > 0; st >>= 1) {
    if (tid < st) { r1[tid] += r1[tid + st]; r2[tid] += r2[tid + st]; }
    __syncthreads();
  }
  float mean = r1[0] * (1.f / 32768.f);
  float var = r2[0] * (1.f / 32768.f) - mean * mean;
  float rstd = rsqrtf(var + 1e-5f);
#pragma unroll 4
  for (int i = 0; i < 32; ++i) {
    int idx = tid + 256 * i;
    int c = g * 32 + (idx >> 8);
    float sc = rstd * w[c], sh = bb[c] - mean * rstd * w[c];
    float4 v = xb4[idx];
    float4 r = {v.x * sc + sh, v.y * sc + sh, v.z * sc + sh, v.w * sc + sh};
    xo4[idx] = r;
  }
}

// ---------------- GEMM: out(M x N) = W(M x K) @ X(K x N), per batch ----------------
// W is pre-converted bf16 -> A-tile staged via async global->LDS b128 copies.
template <typename TB, bool IS_QKV>
__global__ __launch_bounds__(256) void gemm_kernel(
    const __bf16* __restrict__ W, const TB* __restrict__ X,
    const float* __restrict__ bias, const float* __restrict__ res,
    __bf16* __restrict__ out_bf, float* __restrict__ out_f,
    int M, int K, int N, float scale) {
  __shared__ __bf16 At[128 * 32];  // [m][k]
  __shared__ __bf16 Bt[128 * 32];  // [n][k]
  int tid = threadIdx.x, wave = tid >> 5;
  int wm = wave >> 1, wn = wave & 1;
  int b = blockIdx.z;
  int mblk = blockIdx.y * 128, nblk = blockIdx.x * 128;
  const TB* Xb = X + (size_t)b * K * N;

  v8f Z = {0.f, 0.f, 0.f, 0.f, 0.f, 0.f, 0.f, 0.f};
  v8f acc[2][4];
#pragma unroll
  for (int i = 0; i < 2; ++i)
#pragma unroll
    for (int j = 0; j < 4; ++j) acc[i][j] = Z;

  for (int k0 = 0; k0 < K; k0 += 32) {
    __syncthreads();
    // A tile 128x32 bf16: pure copy, async b128 (2 chunks of 8 elems per thread)
#pragma unroll
    for (int i = 0; i < 2; ++i) {
      int idx = tid + 256 * i;           // chunk of 8 bf16
      int m = idx >> 2, kk = (idx & 3) * 8;
      async_copy_b128(&W[(size_t)(mblk + m) * K + k0 + kk], &At[idx * 8]);
    }
    // B tile 32x128 -> Bt[n][k] (transpose): vector global reads, scatter LDS stores
    {
      int kk = tid >> 3;            // 0..31
      int n0 = (tid & 7) * 16;      // 16 n per thread
      const TB* src = &Xb[(size_t)(k0 + kk) * N + nblk + n0];
      if constexpr (sizeof(TB) == 4) {
#pragma unroll
        for (int i = 0; i < 4; ++i) {
          float4 f = ((const float4*)src)[i];
          Bt[(n0 + 4 * i + 0) * 32 + kk] = (__bf16)f.x;
          Bt[(n0 + 4 * i + 1) * 32 + kk] = (__bf16)f.y;
          Bt[(n0 + 4 * i + 2) * 32 + kk] = (__bf16)f.z;
          Bt[(n0 + 4 * i + 3) * 32 + kk] = (__bf16)f.w;
        }
      } else {
        union { uint4 u; __bf16 h[8]; } U[2];
        U[0].u = ((const uint4*)src)[0];
        U[1].u = ((const uint4*)src)[1];
#pragma unroll
        for (int i = 0; i < 16; ++i)
          Bt[(n0 + i) * 32 + kk] = U[i >> 3].h[i & 7];
      }
    }
    wait_async0();
    __syncthreads();
    v16bf af[2], bfr[4];
#pragma unroll
    for (int am = 0; am < 2; ++am)
      af[am] = load_fragA(At + (wm * 32 + am * 16) * 32, 32);
#pragma unroll
    for (int bn = 0; bn < 4; ++bn)
      bfr[bn] = load_fragB(Bt + (wn * 64 + bn * 16) * 32, 32);
#pragma unroll
    for (int am = 0; am < 2; ++am)
#pragma unroll
      for (int bn = 0; bn < 4; ++bn)
        acc[am][bn] = wmma_bf16(af[am], bfr[bn], acc[am][bn]);
  }

  int l = tid & 31, half = l >> 4, col = l & 15;
#pragma unroll
  for (int am = 0; am < 2; ++am)
#pragma unroll
    for (int bn = 0; bn < 4; ++bn)
#pragma unroll
      for (int j = 0; j < 8; ++j) {
        int o = mblk + wm * 32 + am * 16 + j + 8 * half;
        int n = nblk + wn * 64 + bn * 16 + col;
        float v = acc[am][bn][j];
        if constexpr (IS_QKV) {
          v += bias[o];
          if ((o % 192) < 128) v *= scale;  // fold 1/sqrt(sqrt(ch)) into q and k
          out_bf[((size_t)b * M + o) * N + n] = (__bf16)v;
        } else {
          v += bias[o] + res[((size_t)b * M + o) * N + n];
          out_f[((size_t)b * M + o) * N + n] = v;
        }
      }
}

// ---------------- Flash attention: one head x 128 query columns per block ----------------
__global__ __launch_bounds__(256) void attn_kernel(const __bf16* __restrict__ qkv,
                                                   __bf16* __restrict__ hout) {
  __shared__ __bf16 Qt[128 * 64];    // [t][c]
  __shared__ __bf16 Kt[32 * 64];     // [s][c]
  __shared__ __bf16 Pt[8][16 * 32];  // per-wave [t][s]
  int tid = threadIdx.x, wave = tid >> 5, l = tid & 31;
  int half = l >> 4, col = l & 15;
  int tblk = blockIdx.x, bh = blockIdx.y;
  int b = bh >> 4, hd = bh & 15;
  size_t qoff = ((size_t)b * (3 * C_) + hd * (3 * CH_)) * L_;
  const __bf16* qp = qkv + qoff;
  const __bf16* kp = qp + (size_t)CH_ * L_;
  const __bf16* vp = qp + (size_t)(2 * CH_) * L_;
  int t0 = tblk * 128;

  // stage Q block transposed: [t][c]; b128 global reads, scatter LDS stores
#pragma unroll
  for (int i = 0; i < 4; ++i) {
    int idx = tid + 256 * i;       // 1024 chunks of 8
    int c = idx >> 4;              // 0..63
    int t8 = (idx & 15) * 8;
    union { uint4 u; __bf16 h[8]; } U;
    U.u = *(const uint4*)&qp[(size_t)c * L_ + t0 + t8];
#pragma unroll
    for (int j = 0; j < 8; ++j) Qt[(t8 + j) * 64 + c] = U.h[j];
  }
  __syncthreads();
  v16bf qf0 = load_fragB(Qt + (wave * 16) * 64, 64);       // c 0..31
  v16bf qf1 = load_fragB(Qt + (wave * 16) * 64 + 32, 64);  // c 32..63

  v8f Z = {0.f, 0.f, 0.f, 0.f, 0.f, 0.f, 0.f, 0.f};
  v8f O[4];
#pragma unroll
  for (int ct = 0; ct < 4; ++ct) O[ct] = Z;
  float m_run = -1e30f, l_run = 0.f;

  for (int s0 = 0; s0 < L_; s0 += 32) {
    __syncthreads();
    {  // stage K block transposed: [s][c]; one b128 read per thread
      int c = tid >> 2, s8 = (tid & 3) * 8;
      union { uint4 u; __bf16 h[8]; } U;
      U.u = *(const uint4*)&kp[(size_t)c * L_ + s0 + s8];
#pragma unroll
      for (int j = 0; j < 8; ++j) Kt[(s8 + j) * 64 + c] = U.h[j];
    }
    __syncthreads();

    // S^T tiles (s x t): A = k^T, B = q
    v16bf ka0 = load_fragA(Kt, 64);
    v16bf ka0b = load_fragA(Kt + 32, 64);
    v16bf ka1 = load_fragA(Kt + 16 * 64, 64);
    v16bf ka1b = load_fragA(Kt + 16 * 64 + 32, 64);
    v8f sc0 = Z, sc1 = Z;
    sc0 = wmma_bf16(ka0, qf0, sc0);
    sc0 = wmma_bf16(ka0b, qf1, sc0);
    sc1 = wmma_bf16(ka1, qf0, sc1);
    sc1 = wmma_bf16(ka1b, qf1, sc1);

    // online softmax over s (per t-column = per lane pair across half-waves)
    float mt = -1e30f;
#pragma unroll
    for (int j = 0; j < 8; ++j) { mt = fmaxf(mt, sc0[j]); mt = fmaxf(mt, sc1[j]); }
    mt = fmaxf(mt, __shfl_xor(mt, 16, 32));
    float mnew = fmaxf(m_run, mt);
    float alpha = __expf(m_run - mnew);
    float p0[8], p1[8], rs = 0.f;
#pragma unroll
    for (int j = 0; j < 8; ++j) {
      p0[j] = __expf(sc0[j] - mnew);
      p1[j] = __expf(sc1[j] - mnew);
      rs += p0[j] + p1[j];
    }
    rs += __shfl_xor(rs, 16, 32);
    l_run = l_run * alpha + rs;
    m_run = mnew;

    // pack P (bf16) into per-wave LDS as [t][s]
    unsigned* prow = (unsigned*)(&Pt[wave][col * 32]);
#pragma unroll
    for (int i = 0; i < 4; ++i) {
      prow[4 * half + i] = pack_bf16(p0[2 * i], p0[2 * i + 1]);
      prow[8 + 4 * half + i] = pack_bf16(p1[2 * i], p1[2 * i + 1]);
    }
#pragma unroll
    for (int ct = 0; ct < 4; ++ct)
#pragma unroll
      for (int j = 0; j < 8; ++j) O[ct][j] *= alpha;

    // O(c x t) += V(c x s) @ P(s x t); V frags straight from global (s contiguous)
    v16bf pf = load_fragB(&Pt[wave][0], 32);
#pragma unroll
    for (int ct = 0; ct < 4; ++ct) {
      v16bf vf = load_fragA(vp + (size_t)(ct * 16) * L_ + s0, L_);
      O[ct] = wmma_bf16(vf, pf, O[ct]);
    }
  }

  float inv = 1.0f / l_run;
  int t = t0 + wave * 16 + col;
#pragma unroll
  for (int ct = 0; ct < 4; ++ct)
#pragma unroll
    for (int j = 0; j < 8; ++j) {
      int c = ct * 16 + j + 8 * half;
      hout[((size_t)b * C_ + hd * CH_ + c) * L_ + t] = (__bf16)(O[ct][j] * inv);
    }
}

extern "C" void kernel_launch(void* const* d_in, const int* in_sizes, int n_in,
                              void* d_out, int out_size, void* d_ws, size_t ws_size,
                              hipStream_t stream) {
  const float* x = (const float*)d_in[0];
  const float* norm_w = (const float*)d_in[1];
  const float* norm_b = (const float*)d_in[2];
  const float* qkv_w = (const float*)d_in[3];
  const float* qkv_b = (const float*)d_in[4];
  const float* proj_w = (const float*)d_in[5];
  const float* proj_b = (const float*)d_in[6];
  float* out = (float*)d_out;

  const size_t XN_BYTES = (size_t)B_ * C_ * L_ * sizeof(float);        // 32 MB
  const size_t QKV_BYTES = (size_t)B_ * 3 * C_ * L_ * sizeof(__bf16);  // 48 MB
  const size_t H_BYTES = (size_t)B_ * C_ * L_ * sizeof(__bf16);        // 16 MB
  const size_t WQKV_BYTES = (size_t)3 * C_ * C_ * sizeof(__bf16);      // 6 MB
  char* ws = (char*)d_ws;
  float* xn = (float*)ws;
  __bf16* qkvbuf = (__bf16*)(ws + XN_BYTES);
  __bf16* hbuf = (__bf16*)(ws + XN_BYTES + QKV_BYTES);
  __bf16* wqkv = (__bf16*)(ws + XN_BYTES + QKV_BYTES + H_BYTES);
  __bf16* wproj = (__bf16*)(ws + XN_BYTES + QKV_BYTES + H_BYTES + WQKV_BYTES);

  const float scale = 0.35355339059327373f;  // 1/sqrt(sqrt(64))

  cvt_bf16_kernel<<<dim3((3 * C_ * C_) / 1024), dim3(256), 0, stream>>>(qkv_w, wqkv);
  cvt_bf16_kernel<<<dim3((C_ * C_) / 1024), dim3(256), 0, stream>>>(proj_w, wproj);

  groupnorm_kernel<<<dim3(B_ * GROUPS_), dim3(256), 0, stream>>>(x, norm_w, norm_b, xn);

  gemm_kernel<float, true><<<dim3(L_ / 128, (3 * C_) / 128, B_), dim3(256), 0, stream>>>(
      wqkv, xn, qkv_b, nullptr, qkvbuf, nullptr, 3 * C_, C_, L_, scale);

  attn_kernel<<<dim3(L_ / 128, B_ * H_), dim3(256), 0, stream>>>(qkvbuf, hbuf);

  gemm_kernel<__bf16, false><<<dim3(L_ / 128, C_ / 128, B_), dim3(256), 0, stream>>>(
      wproj, hbuf, proj_b, xn, nullptr, out, C_, C_, L_, 1.0f);
}
